// MultiHeadAttention_24773371363846
// MI455X (gfx1250) — compile-verified, matrix operand order checked by reading, not run on hardware
//
#include <hip/hip_runtime.h>
#include <hip/hip_bf16.h>
#include <math.h>

typedef __attribute__((ext_vector_type(16))) _Float16 v16h;
typedef __attribute__((ext_vector_type(8)))  _Float16 v8h;
typedef __attribute__((ext_vector_type(4)))  _Float16 v4h;
typedef __attribute__((ext_vector_type(8)))  float    v8f;
typedef __attribute__((ext_vector_type(4)))  float    v4f;

#define BATCH   2
#define SLEN    2048
#define DMODEL  512
#define HEADS   8
#define DK      64
#define MROWS   (BATCH * SLEN)   /* 4096 */

// ---------------------------------------------------------------------------
// Fragment loader for a row-major matrix with K contiguous (NT-GEMM operand).
// CDNA5 16-bit A-matrix 16x32 layout: lane&15 = row, lane>>4 selects k-half;
// elems 0..7 = K = k0 + 8*kh .. +7, elems 8..15 = K = k0+16+8*kh .. +7.
// B operand of an NT gemm (rows over K) uses the identical mapping (lane&15=col).
// ---------------------------------------------------------------------------
__device__ __forceinline__ v16h load_frag_k16(const _Float16* base, int ld,
                                              int row0, int k0, int lane) {
  int r  = lane & 15;
  int kh = (lane >> 4) & 1;
  const _Float16* p = base + (size_t)(row0 + r) * ld + k0 + kh * 8;
  v8h lo = *(const v8h*)(p);        // K = k0+8kh .. +7
  v8h hi = *(const v8h*)(p + 16);   // K = k0+16+8kh .. +7
  v16h f;
#pragma unroll
  for (int i = 0; i < 8; ++i) { f[i] = lo[i]; f[i + 8] = hi[i]; }
  return f;
}

__device__ __forceinline__ v8f wmma_f16(v16h a, v16h b, v8f c) {
  return __builtin_amdgcn_wmma_f32_16x16x32_f16(false, a, false, b,
                                                (short)0, c, false, false);
}

// ---------------------------------------------------------------------------
// f32 -> f16 conversion (grid-stride)
// ---------------------------------------------------------------------------
__global__ void f32_to_f16_kernel(const float* __restrict__ src,
                                  _Float16* __restrict__ dst, int n) {
  int i = blockIdx.x * blockDim.x + threadIdx.x;
  int stride = gridDim.x * blockDim.x;
  for (; i < n; i += stride) dst[i] = (_Float16)src[i];
}

// ---------------------------------------------------------------------------
// NT GEMM: Y[m,n] = sum_k A[m,k] * Bw[n,k].  M=4096, N=512, K=512.
// One wave per 16x64 tile of C.  Output modes:
//   0: f16, head-split (B,H,S,DK)              (Q, K projections)
//   1: f16, head-split transposed (B,H,DK,S)   (V projection)
//   2: f32, plain row-major (M, DMODEL)        (final output projection)
// ---------------------------------------------------------------------------
__global__ void gemm_nt_kernel(const _Float16* __restrict__ A,
                               const _Float16* __restrict__ Bw,
                               void* __restrict__ outp, int K, int mode) {
  int mt   = blockIdx.x * 16;
  int nt   = blockIdx.y * 64;
  int lane = threadIdx.x & 31;

  v8f acc[4] = {};
  for (int k0 = 0; k0 < K; k0 += 32) {
    v16h a = load_frag_k16(A, K, mt, k0, lane);
#pragma unroll
    for (int j = 0; j < 4; ++j) {
      v16h b = load_frag_k16(Bw, K, nt + j * 16, k0, lane);
      acc[j] = wmma_f16(a, b, acc[j]);
    }
  }

  // C/D layout: VGPR v -> m = v + 8*(lane>>4), n = lane&15
  int col    = lane & 15;
  int rowadd = 8 * (lane >> 4);
#pragma unroll
  for (int j = 0; j < 4; ++j) {
#pragma unroll
    for (int v = 0; v < 8; ++v) {
      int m = mt + v + rowadd;
      int n = nt + j * 16 + col;
      float val = acc[j][v];
      if (mode == 2) {
        ((float*)outp)[(size_t)m * DMODEL + n] = val;
      } else {
        int b = m / SLEN, s = m % SLEN;
        int h = n / DK,   d = n % DK;
        _Float16 hv = (_Float16)val;
        if (mode == 0)
          ((_Float16*)outp)[(((size_t)b * HEADS + h) * SLEN + s) * DK + d] = hv;
        else
          ((_Float16*)outp)[(((size_t)b * HEADS + h) * DK + d) * SLEN + s] = hv;
      }
    }
  }
}

// ---------------------------------------------------------------------------
// Attention: one block (128 thr = 4 waves) per (b, h, 16-row q tile).
// LDS (dynamic): 16 x 2048 f16 scores/probs (64KB) + 16+16 f32 reductions.
// ---------------------------------------------------------------------------
__global__ void attention_kernel(const _Float16* __restrict__ Qh,
                                 const _Float16* __restrict__ Kh,
                                 const _Float16* __restrict__ Vt,
                                 const unsigned char* __restrict__ mask,
                                 float* __restrict__ attn,
                                 _Float16* __restrict__ Ohead) {
  const int tile = blockIdx.x;
  const int qt   = (tile % (SLEN / 16)) * 16;
  const int bh   = tile / (SLEN / 16);
  const int b    = bh / HEADS;
  const int h    = bh % HEADS;

  const _Float16* Q  = Qh + (size_t)bh * SLEN * DK;   // (S, DK) row-major
  const _Float16* Kp = Kh + (size_t)bh * SLEN * DK;   // (S, DK) row-major
  const _Float16* Vp = Vt + (size_t)bh * DK * SLEN;   // (DK, S) row-major
  const unsigned char* mk = mask + (size_t)b * SLEN;
  float* attn_bh = attn + ((size_t)bh * SLEN + qt) * SLEN;

  extern __shared__ __align__(16) char smem_raw[];
  _Float16* sc     = (_Float16*)smem_raw;                         // [16][SLEN]
  float*    rowmax = (float*)(smem_raw + 16 * SLEN * sizeof(_Float16));
  float*    rowsum = rowmax + 16;

  const int tid  = threadIdx.x;   // 0..127
  const int wave = tid >> 5;      // 0..3
  const int lane = tid & 31;

  // ---- Phase 1: scores = (Q K^T)/8, masked, -> f16 LDS -------------------
  v16h aq0 = load_frag_k16(Q, DK, qt, 0,  lane);
  v16h aq1 = load_frag_k16(Q, DK, qt, 32, lane);
  const int ncol   = lane & 15;
  const int rowadd = 8 * (lane >> 4);
  for (int kt = wave * 16; kt < SLEN; kt += 64) {
    v16h b0 = load_frag_k16(Kp, DK, kt, 0,  lane);
    v16h b1 = load_frag_k16(Kp, DK, kt, 32, lane);
    v8f c = {};
    c = wmma_f16(aq0, b0, c);
    c = wmma_f16(aq1, b1, c);
    int k = kt + ncol;
    float addend = mk[k] ? -INFINITY : 0.f;
#pragma unroll
    for (int v = 0; v < 8; ++v) {
      int q = v + rowadd;
      float s = c[v] * 0.125f + addend;   // 1/sqrt(64), -inf if masked
      sc[q * SLEN + k] = (_Float16)s;
    }
  }
  __syncthreads();

  // ---- Phase 2: per-row max & sum (8 lanes per row, v8h scans) -----------
  const int row = tid >> 3;     // 0..15
  const int seg = tid & 7;      // 0..7, each scans 256 elements
  const _Float16* sr = sc + row * SLEN + seg * 256;
  float mx = -INFINITY;
#pragma unroll 4
  for (int i = 0; i < 32; ++i) {
    v8h v = *(const v8h*)(sr + i * 8);
#pragma unroll
    for (int j = 0; j < 8; ++j) mx = fmaxf(mx, (float)v[j]);
  }
#pragma unroll
  for (int off = 4; off >= 1; off >>= 1)
    mx = fmaxf(mx, __shfl_xor(mx, off, 32));

  float sum = 0.f;
  if (mx != -INFINITY) {
#pragma unroll 4
    for (int i = 0; i < 32; ++i) {
      v8h v = *(const v8h*)(sr + i * 8);
#pragma unroll
      for (int j = 0; j < 8; ++j) sum += __expf((float)v[j] - mx);
    }
  }
#pragma unroll
  for (int off = 4; off >= 1; off >>= 1)
    sum += __shfl_xor(sum, off, 32);
  if (seg == 0) { rowmax[row] = mx; rowsum[row] = sum; }
  __syncthreads();

  // ---- Phase 3: normalize; coalesced f32x4 attn write; f16 probs to LDS --
  // Linear sweep: each lane owns a float4 quad; a wave stores 512B contiguous
  // per step (the 268MB attn write is the roofline floor -> must coalesce).
  for (int e = tid * 4; e < 16 * SLEN; e += 128 * 4) {
    int r = e >> 11;            // SLEN == 2048
    int k = e & (SLEN - 1);
    float rm = rowmax[r], rs = rowsum[r];   // LDS broadcast (same r per step)
    bool ok = (rm != -INFINITY) && (rs > 0.f);
    float inv = ok ? 1.f / rs : 0.f;
    v4h sv = *(const v4h*)(sc + r * SLEN + k);
    v4f pv;
    v4h ph;
#pragma unroll
    for (int i = 0; i < 4; ++i) {
      float p = ok ? __expf((float)sv[i] - rm) * inv : 0.f;
      pv[i] = p;
      ph[i] = (_Float16)p;
    }
    *(v4f*)(attn_bh + (size_t)r * SLEN + k) = pv;  // global_store_b128
    *(v4h*)(sc + r * SLEN + k) = ph;               // ds_store_b64
  }
  __syncthreads();

  // ---- Phase 4: O = P * V (K dim = 2048) ---------------------------------
  const int dt = wave * 16;       // each wave owns a 16-wide d sub-tile
  v8f o = {};
  for (int k0 = 0; k0 < SLEN; k0 += 32) {
    v16h pa = load_frag_k16(sc, SLEN, 0,  k0, lane);   // LDS: ds_load_b128
    v16h vb = load_frag_k16(Vp, SLEN, dt, k0, lane);   // V transposed (d,k)
    o = wmma_f16(pa, vb, o);
  }
#pragma unroll
  for (int v = 0; v < 8; ++v) {
    int q   = qt + v + rowadd;
    int col = h * DK + dt + ncol;
    Ohead[((size_t)b * SLEN + q) * DMODEL + col] = (_Float16)o[v];
  }
}

// ---------------------------------------------------------------------------
extern "C" void kernel_launch(void* const* d_in, const int* in_sizes, int n_in,
                              void* d_out, int out_size, void* d_ws, size_t ws_size,
                              hipStream_t stream) {
  (void)in_sizes; (void)n_in; (void)out_size; (void)ws_size;

  const float* query = (const float*)d_in[0];
  const float* key_  = (const float*)d_in[1];
  const float* value = (const float*)d_in[2];
  const unsigned char* mask = (const unsigned char*)d_in[3];  // jnp.bool_ = 1B
  const float* W_Q = (const float*)d_in[4];
  const float* W_K = (const float*)d_in[5];
  const float* W_V = (const float*)d_in[6];
  const float* W_O = (const float*)d_in[7];

  char* ws = (char*)d_ws;
  const size_t MB = 1u << 20;
  _Float16* xq = (_Float16*)(ws + 0 * MB);                 // 4 MB each
  _Float16* xk = (_Float16*)(ws + 4 * MB);
  _Float16* xv = (_Float16*)(ws + 8 * MB);
  _Float16* wq = (_Float16*)(ws + 12 * MB);                // 512 KB each
  _Float16* wk = (_Float16*)(ws + 12 * MB + 512 * 1024);
  _Float16* wv = (_Float16*)(ws + 13 * MB);
  _Float16* wo = (_Float16*)(ws + 13 * MB + 512 * 1024);
  _Float16* Qh = (_Float16*)(ws + 14 * MB);                // (B,H,S,DK) f16
  _Float16* Kh = (_Float16*)(ws + 18 * MB);                // (B,H,S,DK) f16
  _Float16* Vt = (_Float16*)(ws + 22 * MB);                // (B,H,DK,S) f16
  _Float16* Oh = (_Float16*)(ws + 26 * MB);                // (B,S,DMODEL) f16

  float* out  = (float*)d_out;                             // (B,S,D)
  float* attn = out + (size_t)BATCH * SLEN * DMODEL;       // (B,H,S,S)

  const int nX = MROWS * DMODEL;        // 2,097,152
  const int nW = DMODEL * DMODEL;       //   262,144

  f32_to_f16_kernel<<<1024, 256, 0, stream>>>(query, xq, nX);
  f32_to_f16_kernel<<<1024, 256, 0, stream>>>(key_,  xk, nX);
  f32_to_f16_kernel<<<1024, 256, 0, stream>>>(value, xv, nX);
  f32_to_f16_kernel<<<256,  256, 0, stream>>>(W_Q, wq, nW);
  f32_to_f16_kernel<<<256,  256, 0, stream>>>(W_K, wk, nW);
  f32_to_f16_kernel<<<256,  256, 0, stream>>>(W_V, wv, nW);
  f32_to_f16_kernel<<<256,  256, 0, stream>>>(W_O, wo, nW);

  dim3 gg(MROWS / 16, DMODEL / 64);     // 256 x 8 tiles, 1 wave each
  gemm_nt_kernel<<<gg, 32, 0, stream>>>(xq, wq, Qh, DMODEL, 0);
  gemm_nt_kernel<<<gg, 32, 0, stream>>>(xk, wk, Kh, DMODEL, 0);
  gemm_nt_kernel<<<gg, 32, 0, stream>>>(xv, wv, Vt, DMODEL, 1);

  size_t smem = (size_t)16 * SLEN * sizeof(_Float16) + 32 * sizeof(float);
  attention_kernel<<<BATCH * HEADS * (SLEN / 16), 128, smem, stream>>>(
      Qh, Kh, Vt, mask, attn, Oh);

  gemm_nt_kernel<<<gg, 32, 0, stream>>>(Oh, wo, d_out, DMODEL, 2);
}
